// Attention_68813966016768
// MI455X (gfx1250) — matrix operand order repaired and checked by reading, not compiled, not run on hardware
//
#include <hip/hip_runtime.h>
#include <hip/hip_bf16.h>

typedef __attribute__((ext_vector_type(16))) __bf16 v16bf;
typedef __attribute__((ext_vector_type(8)))  float  v8f;
typedef __attribute__((ext_vector_type(4)))  unsigned int v4u;
typedef __attribute__((ext_vector_type(8)))  int v8i;
typedef __attribute__((ext_vector_type(4)))  int v4i;

#define S_LEN  2048
#define DMODEL 1024
#define NHEAD  16
#define DH     64
#define BATCH  2
#define NTOK   (BATCH * S_LEN)              // 4096
#define N_QKV  (3 * DMODEL)                 // 3072
#define BHSD   (BATCH * NHEAD * S_LEN * DH) // 4194304
#define TILE_ELEMS (64 * DH)                // one K/V tile: 4096 bf16 = 8 KB

union Frag {
  v16bf v;
  unsigned int u[8];
  unsigned short s16[16];
};

__device__ __forceinline__ unsigned short f2bf(float f) {
  unsigned int u = __float_as_uint(f);
  u += 0x7FFFu + ((u >> 16) & 1u);   // round-to-nearest-even
  return (unsigned short)(u >> 16);
}

// 16-bit A-matrix 16x32 layout (ISA 7.12.2): per lane (m = lane&15, half = lane>>4),
// VGPR g holds the K pair starting at:
__device__ __forceinline__ int a_kbase(int g, int half) {
  return ((g < 4) ? (2 * g) : (16 + 2 * (g - 4))) + (half ? 8 : 0);
}

// ---------------------------------------------------------------------------
// TDM: DMA a contiguous block of n8 8-byte elements from global to LDS.
// D# per CDNA5 ISA (2-D tensor, groups 2/3 zero).
// ---------------------------------------------------------------------------
#if __has_builtin(__builtin_amdgcn_tensor_load_to_lds)
#define HAVE_TDM 1
__device__ __forceinline__ void tdm_copy_to_lds(unsigned lds_off, const void* gsrc, unsigned n8) {
  unsigned long long ga = (unsigned long long)gsrc;
  v4u g0;
  g0[0] = 1u;                                              // count=1, no gather
  g0[1] = lds_off;                                         // lds_addr
  g0[2] = (unsigned)(ga & 0xFFFFFFFFull);                  // global_addr[31:0]
  g0[3] = (unsigned)((ga >> 32) & 0x01FFFFFFull)           // global_addr[56:32]
          | (2u << 30);                                    // type=2 ("image")
  v8i g1;
  g1[0] = (int)(3u << 16);                                 // data_size=8B
  g1[1] = (int)((n8 & 0xFFFFu) << 16);                     // tensor_dim0[15:0]
  g1[2] = (int)(((n8 >> 16) & 0xFFFFu) | (1u << 16));      // tensor_dim0[31:16], tensor_dim1=1
  g1[3] = (int)((n8 & 0xFFFFu) << 16);                     // tile_dim0 = n8
  g1[4] = 1;                                               // tile_dim1=1, tile_dim2=0
  g1[5] = (int)n8;                                         // tensor_dim0_stride
  g1[6] = 0;
  g1[7] = 0;
  v4i z = {0, 0, 0, 0};
#if __clang_major__ >= 23
  v8i z8 = {0, 0, 0, 0, 0, 0, 0, 0};
  __builtin_amdgcn_tensor_load_to_lds(g0, g1, z, z, z8, 0);
#else
  __builtin_amdgcn_tensor_load_to_lds(g0, g1, z, z, 0);
#endif
}
#else
#define HAVE_TDM 0
#endif

__global__ void cvt_bf16(const float* __restrict__ in, unsigned short* __restrict__ out, int n) {
  int i = blockIdx.x * blockDim.x + threadIdx.x;
  int stride = gridDim.x * blockDim.x;
  for (; i < n; i += stride) out[i] = f2bf(in[i]);
}

// ---------------------------------------------------------------------------
// QKV GEMM: qkv[4096 x 3072] = x_bf @ w_attn_bf + b_attn
// One 32x32 tile per wave (2 A frags x 2 B frags -> 4 WMMA accumulators).
// Scatters q,k,v to head-major bf16 buffers; k,v fp32 into `present`.
// ---------------------------------------------------------------------------
__global__ __launch_bounds__(256) void qkv_gemm(
    const unsigned short* __restrict__ A,    // x bf16 [NTOK][DMODEL]
    const unsigned short* __restrict__ Bw,   // w_attn bf16 [DMODEL][N_QKV]
    const float* __restrict__ bias,          // [N_QKV]
    unsigned short* __restrict__ q_bf,       // [B,H,S,Dh]
    unsigned short* __restrict__ k_bf,
    unsigned short* __restrict__ v_bf,
    float* __restrict__ present) {           // [2,B,H,S,Dh]
  const int lane = threadIdx.x & 31;
  const int waveId = (blockIdx.x * 256 + threadIdx.x) >> 5;
  const int ntn = N_QKV / 32;                // 96
  const int m0 = (waveId / ntn) * 32;
  const int n0 = (waveId % ntn) * 32;
  const int half = lane >> 4;
  const int ml = lane & 15;

  v8f acc[2][2] = {};
  for (int k0 = 0; k0 < DMODEL; k0 += 32) {
    Frag a[2], b[2];
#pragma unroll
    for (int im = 0; im < 2; ++im) {
      const unsigned short* arow = A + (size_t)(m0 + im * 16 + ml) * DMODEL + k0;
#pragma unroll
      for (int g = 0; g < 8; ++g)
        a[im].u[g] = *(const unsigned int*)(arow + a_kbase(g, half));
    }
#pragma unroll
    for (int jn = 0; jn < 2; ++jn) {
      // B layout: lane%16 = N column, elements = K pairs per a_kbase (strided halfwords)
      const unsigned short* brow = Bw + (size_t)k0 * N_QKV + n0 + jn * 16 + ml;
#pragma unroll
      for (int g = 0; g < 8; ++g) {
        b[jn].s16[2 * g]     = brow[(size_t)a_kbase(g, half) * N_QKV];
        b[jn].s16[2 * g + 1] = brow[(size_t)(a_kbase(g, half) + 1) * N_QKV];
      }
    }
#pragma unroll
    for (int im = 0; im < 2; ++im)
#pragma unroll
      for (int jn = 0; jn < 2; ++jn)
        acc[im][jn] = __builtin_amdgcn_wmma_f32_16x16x32_bf16(
            false, a[im].v, false, b[jn].v, (short)0, acc[im][jn], false, false);
  }

#pragma unroll
  for (int jn = 0; jn < 2; ++jn) {
    const int n = n0 + jn * 16 + ml;
    const float bn = bias[n];
    const int part = n / DMODEL;
    const int rem = n - part * DMODEL;
    const int h = rem >> 6;
    const int dh = rem & 63;
#pragma unroll
    for (int im = 0; im < 2; ++im) {
#pragma unroll
      for (int r = 0; r < 8; ++r) {
        int m = m0 + im * 16 + r + (half ? 8 : 0);
        float val = acc[im][jn][r] + bn;
        int bb = m >> 11;                 // / S_LEN
        int s = m & (S_LEN - 1);
        int idx = ((bb * NHEAD + h) * S_LEN + s) * DH + dh;
        if (part == 0) {
          q_bf[idx] = f2bf(val);
        } else if (part == 1) {
          k_bf[idx] = f2bf(val);
          present[idx] = val;
        } else {
          v_bf[idx] = f2bf(val);
          present[(size_t)BHSD + idx] = val;
        }
      }
    }
  }
}

// ---------------------------------------------------------------------------
// Flash attention: 1 WG (8 waves) per 128 query rows per (b,h).
// K/V tiles double-buffered in LDS; Tensor Data Mover DMA for tile kb+1
// overlaps WMMA compute on tile kb.
// ---------------------------------------------------------------------------
__global__ __launch_bounds__(256) void flash_attn(
    const unsigned short* __restrict__ q_bf,
    const unsigned short* __restrict__ k_bf,
    const unsigned short* __restrict__ v_bf,
    unsigned short* __restrict__ a_bf) {     // merged [B,S,D] bf16
  __shared__ unsigned short Kt[2][TILE_ELEMS];  // 16 KB
  __shared__ unsigned short Vt[2][TILE_ELEMS];  // 16 KB
  __shared__ unsigned short Pt[8][16 * DH];     // 16 KB (per-wave P scratch)

  const int tid = threadIdx.x;
  const int lane = tid & 31;
  const int wave = tid >> 5;
  const int half = lane >> 4;
  const int ml = lane & 15;
  const int q0 = blockIdx.x * 128;
  const int bh = blockIdx.y;                 // 0..B*H-1
  const int qrow0 = q0 + wave * 16;

  const unsigned kt_off = (unsigned)(unsigned long long)(void*)&Kt[0][0];
  const unsigned vt_off = (unsigned)(unsigned long long)(void*)&Vt[0][0];
  const unsigned short* kv_base_k = k_bf + (size_t)bh * S_LEN * DH;
  const unsigned short* kv_base_v = v_bf + (size_t)bh * S_LEN * DH;

  // Q fragments: Dh=64 -> two 32-wide K chunks, loaded once
  Frag qa[2];
  const unsigned short* qbase = q_bf + ((size_t)bh * S_LEN + (qrow0 + ml)) * DH;
#pragma unroll
  for (int c = 0; c < 2; ++c)
#pragma unroll
    for (int g = 0; g < 8; ++g)
      qa[c].u[g] = *(const unsigned int*)(qbase + c * 32 + a_kbase(g, half));

  v8f o[4] = {};
  float mi[8], li[8];
#pragma unroll
  for (int r = 0; r < 8; ++r) { mi[r] = -1e30f; li[r] = 0.f; }

  const int nkb = q0 / 64 + 2;               // causal: keys <= q0+127

  // prologue: kick off DMA for tile 0
#if HAVE_TDM
  if (wave == 0) {                           // TDM ignores EXEC; uniform branch
    tdm_copy_to_lds(kt_off, kv_base_k, (TILE_ELEMS * 2) / 8);
    tdm_copy_to_lds(vt_off, kv_base_v, (TILE_ELEMS * 2) / 8);
  }
#else
  {
    unsigned kl = kt_off + tid * 16, vl = vt_off + tid * 16;
    unsigned long long kg = (unsigned long long)kv_base_k + tid * 16;
    unsigned long long vg = (unsigned long long)kv_base_v + tid * 16;
    asm volatile("global_load_async_to_lds_b128 %0, %1, off" :: "v"(kl), "v"(kg) : "memory");
    asm volatile("global_load_async_to_lds_b128 %0, %1, off offset:4096" :: "v"(kl), "v"(kg) : "memory");
    asm volatile("global_load_async_to_lds_b128 %0, %1, off" :: "v"(vl), "v"(vg) : "memory");
    asm volatile("global_load_async_to_lds_b128 %0, %1, off offset:4096" :: "v"(vl), "v"(vg) : "memory");
  }
#endif

  for (int kb = 0; kb < nkb; ++kb) {
    const int key0 = kb * 64;
    const int buf = kb & 1;

    // tile kb DMA complete -> visible to all waves
#if HAVE_TDM
    if (wave == 0) __builtin_amdgcn_s_wait_tensorcnt(0);
#else
    asm volatile("s_wait_asynccnt 0x0" ::: "memory");
#endif
    __syncthreads();

    // issue DMA for tile kb+1 into the other buffer; overlaps compute below.
    // (the barrier above ordered it after all reads of that buffer in kb-1)
    if (kb + 1 < nkb) {
      const int nbuf = buf ^ 1;
      const unsigned short* ksrc = kv_base_k + (size_t)(key0 + 64) * DH;
      const unsigned short* vsrc = kv_base_v + (size_t)(key0 + 64) * DH;
#if HAVE_TDM
      if (wave == 0) {
        tdm_copy_to_lds(kt_off + nbuf * TILE_ELEMS * 2, ksrc, (TILE_ELEMS * 2) / 8);
        tdm_copy_to_lds(vt_off + nbuf * TILE_ELEMS * 2, vsrc, (TILE_ELEMS * 2) / 8);
      }
#else
      {
        unsigned kl = kt_off + nbuf * TILE_ELEMS * 2 + tid * 16;
        unsigned vl = vt_off + nbuf * TILE_ELEMS * 2 + tid * 16;
        unsigned long long kg = (unsigned long long)ksrc + tid * 16;
        unsigned long long vg = (unsigned long long)vsrc + tid * 16;
        asm volatile("global_load_async_to_lds_b128 %0, %1, off" :: "v"(kl), "v"(kg) : "memory");
        asm volatile("global_load_async_to_lds_b128 %0, %1, off offset:4096" :: "v"(kl), "v"(kg) : "memory");
        asm volatile("global_load_async_to_lds_b128 %0, %1, off" :: "v"(vl), "v"(vg) : "memory");
        asm volatile("global_load_async_to_lds_b128 %0, %1, off offset:4096" :: "v"(vl), "v"(vg) : "memory");
      }
#endif
    }

    const unsigned short* KtB = Kt[buf];
    const unsigned short* VtB = Vt[buf];

    // S = (Q K^T) / 8 with causal mask
    float p[4][8];
#pragma unroll
    for (int t = 0; t < 4; ++t) {
      v8f st = {};
#pragma unroll
      for (int c = 0; c < 2; ++c) {
        Frag bfr;
        const int kd = c * 32;               // B lane%16 = key column, elements = Dh pairs
#pragma unroll
        for (int g = 0; g < 8; ++g)
          bfr.u[g] = *(const unsigned int*)(&KtB[(t * 16 + ml) * DH + kd + a_kbase(g, half)]);
        st = __builtin_amdgcn_wmma_f32_16x16x32_bf16(false, qa[c].v, false, bfr.v,
                                                     (short)0, st, false, false);
      }
#pragma unroll
      for (int r = 0; r < 8; ++r) {
        int m = qrow0 + r + (half ? 8 : 0);
        int key = key0 + t * 16 + ml;
        float sc = st[r] * 0.125f;           // 1/sqrt(64)
        p[t][r] = (key > m) ? -1e10f : sc;   // matches w*b - 1e10*(1-b)
      }
    }

    // online softmax (row reductions within each 16-lane half)
#pragma unroll
    for (int r = 0; r < 8; ++r) {
      float rmax = fmaxf(fmaxf(p[0][r], p[1][r]), fmaxf(p[2][r], p[3][r]));
      rmax = fmaxf(rmax, __shfl_xor(rmax, 1, 32));
      rmax = fmaxf(rmax, __shfl_xor(rmax, 2, 32));
      rmax = fmaxf(rmax, __shfl_xor(rmax, 4, 32));
      rmax = fmaxf(rmax, __shfl_xor(rmax, 8, 32));
      float mnew = fmaxf(mi[r], rmax);
      float scale = __expf(mi[r] - mnew);
      float rsum = 0.f;
#pragma unroll
      for (int t = 0; t < 4; ++t) {
        p[t][r] = __expf(p[t][r] - mnew);
        rsum += p[t][r];
      }
      rsum += __shfl_xor(rsum, 1, 32);
      rsum += __shfl_xor(rsum, 2, 32);
      rsum += __shfl_xor(rsum, 4, 32);
      rsum += __shfl_xor(rsum, 8, 32);
      li[r] = li[r] * scale + rsum;
      mi[r] = mnew;
#pragma unroll
      for (int t = 0; t < 4; ++t) o[t][r] *= scale;
    }

    // P (C/D layout) -> per-wave LDS row-major 16x64 bf16
#pragma unroll
    for (int t = 0; t < 4; ++t)
#pragma unroll
      for (int r = 0; r < 8; ++r)
        Pt[wave][(r + (half ? 8 : 0)) * DH + t * 16 + ml] = f2bf(p[t][r]);

    // O += P @ V
#pragma unroll
    for (int c = 0; c < 2; ++c) {
      Frag pa;
#pragma unroll
      for (int g = 0; g < 8; ++g)
        pa.u[g] = *(const unsigned int*)(&Pt[wave][ml * DH + c * 32 + a_kbase(g, half)]);
#pragma unroll
      for (int t = 0; t < 4; ++t) {
        Frag vb;
#pragma unroll
        for (int g = 0; g < 8; ++g) {
          vb.s16[2 * g]     = VtB[(c * 32 + a_kbase(g, half)) * DH + t * 16 + ml];
          vb.s16[2 * g + 1] = VtB[(c * 32 + a_kbase(g, half) + 1) * DH + t * 16 + ml];
        }
        o[t] = __builtin_amdgcn_wmma_f32_16x16x32_bf16(false, pa.v, false, vb.v,
                                                       (short)0, o[t], false, false);
      }
    }
  }

  // normalize and write merged heads: a_bf[b][s][h*64+dh]
  const int bb = bh >> 4;
  const int h = bh & 15;
#pragma unroll
  for (int r = 0; r < 8; ++r) {
    float inv = 1.f / li[r];
    int s = qrow0 + r + (half ? 8 : 0);
#pragma unroll
    for (int t = 0; t < 4; ++t) {
      int dh = t * 16 + ml;
      a_bf[(size_t)(bb * S_LEN + s) * DMODEL + h * DH + dh] = f2bf(o[t][r] * inv);
    }
  }
}

// ---------------------------------------------------------------------------
// Output projection: out[4096 x 1024] = a_bf @ w_proj_bf + b_proj (fp32 out)
// One 32x32 tile per wave.
// ---------------------------------------------------------------------------
__global__ __launch_bounds__(256) void proj_gemm(
    const unsigned short* __restrict__ A,    // [NTOK][DMODEL] bf16
    const unsigned short* __restrict__ Bw,   // [DMODEL][DMODEL] bf16
    const float* __restrict__ bias,          // [DMODEL]
    float* __restrict__ C) {                 // [NTOK][DMODEL] fp32
  const int lane = threadIdx.x & 31;
  const int waveId = (blockIdx.x * 256 + threadIdx.x) >> 5;
  const int ntn = DMODEL / 32;               // 32
  const int m0 = (waveId / ntn) * 32;
  const int n0 = (waveId % ntn) * 32;
  const int half = lane >> 4;
  const int ml = lane & 15;

  v8f acc[2][2] = {};
  for (int k0 = 0; k0 < DMODEL; k0 += 32) {
    Frag a[2], b[2];
#pragma unroll
    for (int im = 0; im < 2; ++im) {
      const unsigned short* arow = A + (size_t)(m0 + im * 16 + ml) * DMODEL + k0;
#pragma unroll
      for (int g = 0; g < 8; ++g)
        a[im].u[g] = *(const unsigned int*)(arow + a_kbase(g, half));
    }
#pragma unroll
    for (int jn = 0; jn < 2; ++jn) {
      const unsigned short* brow = Bw + (size_t)k0 * DMODEL + n0 + jn * 16 + ml;
#pragma unroll
      for (int g = 0; g < 8; ++g) {
        b[jn].s16[2 * g]     = brow[(size_t)a_kbase(g, half) * DMODEL];
        b[jn].s16[2 * g + 1] = brow[(size_t)(a_kbase(g, half) + 1) * DMODEL];
      }
    }
#pragma unroll
    for (int im = 0; im < 2; ++im)
#pragma unroll
      for (int jn = 0; jn < 2; ++jn)
        acc[im][jn] = __builtin_amdgcn_wmma_f32_16x16x32_bf16(
            false, a[im].v, false, b[jn].v, (short)0, acc[im][jn], false, false);
  }
#pragma unroll
  for (int jn = 0; jn < 2; ++jn) {
    const int n = n0 + jn * 16 + ml;
    const float bn = bias[n];
#pragma unroll
    for (int im = 0; im < 2; ++im)
#pragma unroll
      for (int r = 0; r < 8; ++r)
        C[(size_t)(m0 + im * 16 + r + (half ? 8 : 0)) * DMODEL + n] = acc[im][jn][r] + bn;
  }
}

extern "C" void kernel_launch(void* const* d_in, const int* in_sizes, int n_in,
                              void* d_out, int out_size, void* d_ws, size_t ws_size,
                              hipStream_t stream) {
  (void)in_sizes; (void)n_in; (void)out_size; (void)ws_size;
  const float* x      = (const float*)d_in[0];
  const float* w_attn = (const float*)d_in[1];
  const float* b_attn = (const float*)d_in[2];
  const float* w_proj = (const float*)d_in[3];
  const float* b_proj = (const float*)d_in[4];

  float* out     = (float*)d_out;                       // a: [B,S,D]
  float* present = out + (size_t)NTOK * DMODEL;         // [2,B,H,S,Dh]

  unsigned short* ws = (unsigned short*)d_ws;
  size_t off = 0;
  unsigned short* x_bf   = ws + off; off += (size_t)NTOK * DMODEL;   // 4M
  unsigned short* wat_bf = ws + off; off += (size_t)DMODEL * N_QKV;  // 3M
  unsigned short* wpr_bf = ws + off; off += (size_t)DMODEL * DMODEL; // 1M
  unsigned short* q_bf   = ws + off; off += (size_t)BHSD;
  unsigned short* k_bf   = ws + off; off += (size_t)BHSD;
  unsigned short* v_bf   = ws + off; off += (size_t)BHSD;
  unsigned short* a_bf   = ws + off;                                 // 4M

  cvt_bf16<<<2048, 256, 0, stream>>>(x, x_bf, NTOK * DMODEL);
  cvt_bf16<<<2048, 256, 0, stream>>>(w_attn, wat_bf, DMODEL * N_QKV);
  cvt_bf16<<<1024, 256, 0, stream>>>(w_proj, wpr_bf, DMODEL * DMODEL);

  // 128x96 32x32 tiles, 8 waves / block -> 1536 blocks
  qkv_gemm<<<(NTOK / 32) * (N_QKV / 32) / 8, 256, 0, stream>>>(
      x_bf, wat_bf, b_attn, q_bf, k_bf, v_bf, present);

  flash_attn<<<dim3(S_LEN / 128, BATCH * NHEAD), 256, 0, stream>>>(
      q_bf, k_bf, v_bf, a_bf);

  // 128x32 tiles -> 512 blocks
  proj_gemm<<<(NTOK / 32) * (DMODEL / 32) / 8, 256, 0, stream>>>(
      a_bf, wpr_bf, b_proj, out);
}
